// GraphFusionModel_40157944218453
// MI455X (gfx1250) — matrix operand order repaired and checked by reading, not compiled, not run on hardware
//
#include <hip/hip_runtime.h>
#include <math.h>

#define NT 768
#define NS 3
#define NN (NT*NS)          // 2304
#define DIMX 128
#define NH 4
#define DKX 32
#define FF 512
#define SROW 2320           // padded LDS score-row stride (floats)

typedef _Float16 v16h __attribute__((ext_vector_type(16)));
typedef _Float16 v8h  __attribute__((ext_vector_type(8)));
typedef float    v8f  __attribute__((ext_vector_type(8)));

__device__ __forceinline__ v16h cat8(v8h lo, v8h hi) {
    return __builtin_shufflevector(lo, hi, 0,1,2,3,4,5,6,7,8,9,10,11,12,13,14,15);
}
__device__ __forceinline__ v8f wmma16(v16h a, v16h b, v8f c) {
    return __builtin_amdgcn_wmma_f32_16x16x32_f16(false, a, false, b, (short)0, c, false, false);
}
__device__ __forceinline__ float gelu_exact(float x) {
    return 0.5f * x * (1.0f + erff(x * 0.70710678118654752f));
}

// ---------------- embedding ----------------
__global__ __launch_bounds__(256)
void k_embed(const float* __restrict__ obs, const float* __restrict__ conf,
             const float* __restrict__ pw, const float* __restrict__ pb,
             const float* __restrict__ temb, const float* __restrict__ semb,
             float* __restrict__ x, _Float16* __restrict__ xh) {
    int idx = blockIdx.x * 256 + threadIdx.x;
    if (idx >= NN * DIMX) return;
    int n = idx >> 7, d = idx & 127;
    const float* o = obs + n * 7;
    float nr[8] = { o[0], o[1], o[2], o[3], o[4], o[5], o[6], conf[n] };
    float acc = pb[d];
    #pragma unroll
    for (int k = 0; k < 8; k++) acc += nr[k] * pw[d * 8 + k];
    int t = n / NS, s = n % NS;
    int tc = t < 255 ? t : 255;
    acc += temb[tc * DIMX + d] + semb[s * DIMX + d];
    x[idx] = acc;
    xh[idx] = (_Float16)acc;
}

// ---------------- per-layer prep: f16 weights + per-node edge aux ----------------
__global__ __launch_bounds__(256)
void k_prep(const float* __restrict__ Wq, const float* __restrict__ Wk, const float* __restrict__ Wv,
            const float* __restrict__ W1, const float* __restrict__ W2,
            const float* __restrict__ obs, const float* __restrict__ conf,
            const float* __restrict__ ew, int layer,
            _Float16* __restrict__ wqh, _Float16* __restrict__ wkh, _Float16* __restrict__ wvh,
            _Float16* __restrict__ w1h, _Float16* __restrict__ w2h,
            float* __restrict__ aux) {
    int idx = blockIdx.x * 256 + threadIdx.x;
    if (idx < 49152) {
        int m = idx / 16384, r = idx % 16384;
        const float* src = (m == 0) ? Wq : (m == 1) ? Wk : Wv;
        _Float16* dst = (m == 0) ? wqh : (m == 1) ? wkh : wvh;
        dst[r] = (_Float16)src[layer * 16384 + r];
    } else if (idx < 114688) {
        int r = idx - 49152;
        w1h[r] = (_Float16)W1[layer * (FF * DIMX) + r];
    } else if (idx < 180224) {
        int r = idx - 114688;
        w2h[r] = (_Float16)W2[layer * (DIMX * FF) + r];
    }
    if (idx < NN) {
        int n = idx;
        const float* o = obs + n * 7;
        float cf = conf[n];
        float* a = aux + n * 8;
        a[0] = o[0]; a[1] = o[1]; a[2] = o[2];
        a[3] = (float)(n / NS);
        #pragma unroll
        for (int h = 0; h < NH; h++) {
            const float* w = ew + (layer * NH + h) * 9;
            a[4 + h] = w[0]*o[0] + w[1]*o[1] + w[2]*o[2]
                     + w[4]*o[3] + w[5]*o[4] + w[6]*o[5] + w[7]*cf;
        }
    }
}

// ---------------- QKV projection (WMMA) ----------------
__global__ __launch_bounds__(256)
void k_qkv(const _Float16* __restrict__ xh,
           const _Float16* __restrict__ wqh, const _Float16* __restrict__ wkh, const _Float16* __restrict__ wvh,
           const float* __restrict__ qb, const float* __restrict__ kb, const float* __restrict__ vb,
           _Float16* __restrict__ Qh, _Float16* __restrict__ Kh, _Float16* __restrict__ VT) {
    int wave = threadIdx.x >> 5, lane = threadIdx.x & 31;
    int task = blockIdx.x * 8 + wave;          // 3456 tasks
    int rt = task / 24, rem = task % 24, mat = rem >> 3, ot = rem & 7;
    int lq = lane & 15;
    int off0 = (lane < 16) ? 0 : 8;
    int koff0 = (lane < 16) ? 0 : 16;
    int row0 = rt * 16 + lq;
    v16h A[4];
    #pragma unroll
    for (int kc = 0; kc < 4; kc++) {
        const _Float16* p = xh + row0 * DIMX + kc * 32;
        A[kc] = cat8(*(const v8h*)(p + off0), *(const v8h*)(p + 16 + off0));
    }
    const _Float16* W = (mat == 0) ? wqh : (mat == 1) ? wkh : wvh;
    const float* bp   = (mat == 0) ? qb  : (mat == 1) ? kb  : vb;
    int o = ot * 16 + lq;
    v8f acc = {};
    #pragma unroll
    for (int kc = 0; kc < 4; kc++) {
        v16h B = cat8(*(const v8h*)(W + o * DIMX + kc * 32 + off0), *(const v8h*)(W + o * DIMX + kc * 32 + 16 + off0));
        acc = wmma16(A[kc], B, acc);
    }
    float bias = bp[o];
    int hbase = (lane < 16) ? 0 : 8;
    if (mat == 0) {
        #pragma unroll
        for (int v = 0; v < 8; v++) Qh[(rt*16 + v + hbase) * DIMX + o] = (_Float16)(acc[v] + bias);
    } else if (mat == 1) {
        #pragma unroll
        for (int v = 0; v < 8; v++) Kh[(rt*16 + v + hbase) * DIMX + o] = (_Float16)(acc[v] + bias);
    } else {
        #pragma unroll
        for (int v = 0; v < 8; v++) VT[o * NN + (rt*16 + v + hbase)] = (_Float16)(acc[v] + bias);
    }
}

// ---------------- attention: WMMA scores + fused edge bias + LDS softmax + WMMA P·V ----------------
__global__ __launch_bounds__(256)
void k_attn(const _Float16* __restrict__ Qh, const _Float16* __restrict__ Kh,
            const _Float16* __restrict__ VT, const float* __restrict__ aux,
            const float* __restrict__ ew, const float* __restrict__ eb, int layer,
            float* __restrict__ attout) {
    extern __shared__ char smem[];
    float* S      = (float*)smem;                       // 16 x SROW
    float* qn     = (float*)(smem + 16 * SROW * 4);     // 16 x 8
    float* red    = qn + 128;                           // 16 x 16
    float* rowmax = red + 256;                          // 16
    float* rowsum = rowmax + 16;                        // 16
    float* Oacc   = rowsum + 16;                        // 16 x 32

    int qt = blockIdx.x >> 2;
    int h  = blockIdx.x & 3;
    int tid = threadIdx.x, wave = tid >> 5, lane = tid & 31;
    int qbase = qt * 16;
    int lq = lane & 15;
    int off0  = (lane < 16) ? 0 : 8;
    int koff0 = (lane < 16) ? 0 : 16;

    if (tid < 128) qn[tid] = aux[qbase * 8 + tid];
    for (int i = tid; i < 512; i += 256) Oacc[i] = 0.0f;
    __syncthreads();

    const float* w = ew + (layer * NH + h) * 9;
    float w3 = w[3];
    float w8 = w[8] * (1.0f / 768.0f);
    float bh = eb[layer * NH + h];

    // Q operand (A), fixed per block
    const _Float16* qrow = Qh + (qbase + lq) * DIMX + h * DKX;
    v16h Aq = cat8(*(const v8h*)(qrow + off0), *(const v8h*)(qrow + 16 + off0));

    // ---- score phase ----
    for (int kt = wave; kt < NN / 16; kt += 8) {
        int j = kt * 16 + lq;
        v16h B = cat8(*(const v8h*)(Kh + j * DIMX + h * DKX + off0), *(const v8h*)(Kh + j * DIMX + h * DKX + 16 + off0));
        v8f c = {};
        c = wmma16(Aq, B, c);
        const float4 ja = *(const float4*)(aux + j * 8);
        float jl = aux[j * 8 + 4 + h];
        int hbase = (lane < 16) ? 0 : 8;
        #pragma unroll
        for (int v = 0; v < 8; v++) {
            int ql = v + hbase;
            float4 ia = *(const float4*)(qn + ql * 8);
            float il = qn[ql * 8 + 4 + h];
            float dx = ia.x - ja.x, dy = ia.y - ja.y, dz = ia.z - ja.z;
            float dist = sqrtf(dx*dx + dy*dy + dz*dz + 1e-12f);
            float dt = fabsf(ia.w - ja.w);
            float bias = il - jl + w3 * dist + w8 * dt + bh;
            S[ql * SROW + j] = c[v] * 0.17677669529663687f + bias;
        }
    }
    __syncthreads();

    // ---- softmax (rows in LDS) ----
    int r = tid >> 4, cg = tid & 15;
    float m = -1e30f;
    for (int j = cg; j < NN; j += 16) m = fmaxf(m, S[r * SROW + j]);
    red[r * 16 + cg] = m;
    __syncthreads();
    if (tid < 16) {
        float mm = -1e30f;
        #pragma unroll
        for (int i = 0; i < 16; i++) mm = fmaxf(mm, red[tid * 16 + i]);
        rowmax[tid] = mm;
    }
    __syncthreads();
    float mrow = rowmax[r];
    float sum = 0.0f;
    for (int j = cg; j < NN; j += 16) {
        float p = __expf(S[r * SROW + j] - mrow);
        S[r * SROW + j] = p;
        sum += p;
    }
    red[r * 16 + cg] = sum;
    __syncthreads();
    if (tid < 16) {
        float ss = 0.0f;
        #pragma unroll
        for (int i = 0; i < 16; i++) ss += red[tid * 16 + i];
        rowsum[tid] = ss;
    }
    __syncthreads();

    // ---- P @ V phase ----
    v8f o0 = {}, o1 = {};
    const float* srow = S + lq * SROW;
    for (int ch = wave; ch < NN / 32; ch += 8) {
        int c0 = ch * 32;
        v16h Ap;
        #pragma unroll
        for (int e = 0; e < 4; e++) {
            float2 p  = *(const float2*)(srow + c0 + off0 + 2 * e);
            float2 p2 = *(const float2*)(srow + c0 + 16 + off0 + 2 * e);
            Ap[2*e]     = (_Float16)p.x;  Ap[2*e + 1] = (_Float16)p.y;
            Ap[8 + 2*e] = (_Float16)p2.x; Ap[9 + 2*e] = (_Float16)p2.y;
        }
        const _Float16* vtb = VT + (h * DKX + lq) * NN + c0;
        v16h B0 = cat8(*(const v8h*)(vtb + off0), *(const v8h*)(vtb + 16 + off0));
        v16h B1 = cat8(*(const v8h*)(vtb + 16 * NN + off0), *(const v8h*)(vtb + 16 * NN + 16 + off0));
        o0 = wmma16(Ap, B0, o0);
        o1 = wmma16(Ap, B1, o1);
    }
    int hbase = (lane < 16) ? 0 : 8;
    #pragma unroll
    for (int v = 0; v < 8; v++) {
        atomicAdd(&Oacc[(v + hbase) * 32 + lq], o0[v]);
        atomicAdd(&Oacc[(v + hbase) * 32 + 16 + lq], o1[v]);
    }
    __syncthreads();
    for (int i = tid; i < 512; i += 256) {
        int ql = i >> 5, dk = i & 31;
        attout[(qbase + ql) * DIMX + h * DKX + dk] = Oacc[i] / rowsum[ql];
    }
}

// ---------------- residual add + LayerNorm ----------------
__global__ __launch_bounds__(128)
void k_addln(const float* __restrict__ xin, const float* __restrict__ add,
             const float* __restrict__ g, const float* __restrict__ b,
             float* __restrict__ xout, _Float16* __restrict__ xh) {
    __shared__ float red[128];
    int row = blockIdx.x, d = threadIdx.x;
    float v = xin[row * DIMX + d];
    if (add) v += add[row * DIMX + d];
    red[d] = v;
    __syncthreads();
    for (int s = 64; s > 0; s >>= 1) { if (d < s) red[d] += red[d + s]; __syncthreads(); }
    float mean = red[0] * (1.0f / 128.0f);
    __syncthreads();
    float dv = v - mean;
    red[d] = dv * dv;
    __syncthreads();
    for (int s = 64; s > 0; s >>= 1) { if (d < s) red[d] += red[d + s]; __syncthreads(); }
    float var = red[0] * (1.0f / 128.0f);
    float y = dv * rsqrtf(var + 1e-5f) * g[d] + b[d];
    xout[row * DIMX + d] = y;
    xh[row * DIMX + d] = (_Float16)y;
}

// ---------------- FFN1 (WMMA + exact GELU) ----------------
__global__ __launch_bounds__(256)
void k_ffn1(const _Float16* __restrict__ xh, const _Float16* __restrict__ w1h,
            const float* __restrict__ b1, _Float16* __restrict__ g16) {
    int wave = threadIdx.x >> 5, lane = threadIdx.x & 31;
    int task = blockIdx.x * 8 + wave;          // 4608 tasks
    int rt = task >> 5, ot = task & 31;
    int lq = lane & 15;
    int off0 = (lane < 16) ? 0 : 8;
    int koff0 = (lane < 16) ? 0 : 16;
    int row0 = rt * 16 + lq;
    int o = ot * 16 + lq;
    v8f acc = {};
    #pragma unroll
    for (int kc = 0; kc < 4; kc++) {
        const _Float16* p = xh + row0 * DIMX + kc * 32;
        v16h A = cat8(*(const v8h*)(p + off0), *(const v8h*)(p + 16 + off0));
        v16h B = cat8(*(const v8h*)(w1h + o * DIMX + kc * 32 + off0), *(const v8h*)(w1h + o * DIMX + kc * 32 + 16 + off0));
        acc = wmma16(A, B, acc);
    }
    float bias = b1[o];
    int hbase = (lane < 16) ? 0 : 8;
    #pragma unroll
    for (int v = 0; v < 8; v++) {
        int row = rt * 16 + v + hbase;
        g16[row * FF + o] = (_Float16)gelu_exact(acc[v] + bias);
    }
}

// ---------------- FFN2 (WMMA + bias + residual) ----------------
__global__ __launch_bounds__(256)
void k_ffn2(const _Float16* __restrict__ g16, const _Float16* __restrict__ w2h,
            const float* __restrict__ b2, const float* __restrict__ xres,
            float* __restrict__ y) {
    int wave = threadIdx.x >> 5, lane = threadIdx.x & 31;
    int task = blockIdx.x * 8 + wave;          // 1152 tasks
    int rt = task >> 3, ot = task & 7;
    int lq = lane & 15;
    int off0 = (lane < 16) ? 0 : 8;
    int koff0 = (lane < 16) ? 0 : 16;
    int row0 = rt * 16 + lq;
    int o = ot * 16 + lq;
    v8f acc = {};
    #pragma unroll
    for (int kc = 0; kc < 16; kc++) {
        const _Float16* p = g16 + row0 * FF + kc * 32;
        v16h A = cat8(*(const v8h*)(p + off0), *(const v8h*)(p + 16 + off0));
        v16h B = cat8(*(const v8h*)(w2h + o * FF + kc * 32 + off0), *(const v8h*)(w2h + o * FF + kc * 32 + 16 + off0));
        acc = wmma16(A, B, acc);
    }
    float bias = b2[o];
    int hbase = (lane < 16) ? 0 : 8;
    #pragma unroll
    for (int v = 0; v < 8; v++) {
        int row = rt * 16 + v + hbase;
        y[row * DIMX + o] = acc[v] + bias + xres[row * DIMX + o];
    }
}

// ---------------- head ----------------
__global__ __launch_bounds__(256)
void k_fuse(const float* __restrict__ x, float* __restrict__ fused) {
    int idx = blockIdx.x * 256 + threadIdx.x;
    if (idx >= NT * DIMX) return;
    int t = idx >> 7, d = idx & 127;
    float s = x[(t*3+0)*DIMX + d] + x[(t*3+1)*DIMX + d] + x[(t*3+2)*DIMX + d];
    fused[idx] = s * (1.0f / 3.0f);
}

__global__ __launch_bounds__(256)
void k_head1(const float* __restrict__ fused, const float* __restrict__ w,
             const float* __restrict__ b, float* __restrict__ h1) {
    int idx = blockIdx.x * 256 + threadIdx.x;
    if (idx >= NT * DIMX) return;
    int t = idx >> 7, d = idx & 127;
    float acc = b[d];
    const float* fr = fused + t * DIMX;
    const float* wr = w + d * DIMX;
    #pragma unroll 8
    for (int k = 0; k < DIMX; k++) acc += fr[k] * wr[k];
    h1[idx] = gelu_exact(acc);
}

__global__ __launch_bounds__(256)
void k_head2(const float* __restrict__ h1, const float* __restrict__ w,
             const float* __restrict__ b, float* __restrict__ out) {
    int idx = blockIdx.x * 256 + threadIdx.x;
    if (idx >= NT * 3) return;
    int t = idx / 3, c = idx % 3;
    float acc = b[c];
    const float* hr = h1 + t * DIMX;
    const float* wr = w + c * DIMX;
    #pragma unroll 8
    for (int k = 0; k < DIMX; k++) acc += hr[k] * wr[k];
    out[idx] = acc;
}

extern "C" void kernel_launch(void* const* d_in, const int* in_sizes, int n_in,
                              void* d_out, int out_size, void* d_ws, size_t ws_size,
                              hipStream_t stream) {
    const float* traj_obs  = (const float*)d_in[0];
    const float* traj_conf = (const float*)d_in[1];
    const float* node_pw   = (const float*)d_in[2];
    const float* node_pb   = (const float*)d_in[3];
    const float* time_emb  = (const float*)d_in[4];
    const float* src_emb   = (const float*)d_in[5];
    const float* Wq_w = (const float*)d_in[6];
    const float* Wq_b = (const float*)d_in[7];
    const float* Wk_w = (const float*)d_in[8];
    const float* Wk_b = (const float*)d_in[9];
    const float* Wv_w = (const float*)d_in[10];
    const float* Wv_b = (const float*)d_in[11];
    const float* edge_w = (const float*)d_in[12];
    const float* edge_b = (const float*)d_in[13];
    const float* ln1_g = (const float*)d_in[14];
    const float* ln1_b = (const float*)d_in[15];
    const float* ln2_g = (const float*)d_in[16];
    const float* ln2_b = (const float*)d_in[17];
    const float* ffn1_w = (const float*)d_in[18];
    const float* ffn1_b = (const float*)d_in[19];
    const float* ffn2_w = (const float*)d_in[20];
    const float* ffn2_b = (const float*)d_in[21];
    const float* out1_w = (const float*)d_in[22];
    const float* out1_b = (const float*)d_in[23];
    const float* out2_w = (const float*)d_in[24];
    const float* out2_b = (const float*)d_in[25];

    char* ws = (char*)d_ws;
    float*    x     = (float*)(ws + 0);
    _Float16* xh    = (_Float16*)(ws + 1179648);
    _Float16* Qh    = (_Float16*)(ws + 1769472);
    _Float16* Kh    = (_Float16*)(ws + 2359296);
    _Float16* VT    = (_Float16*)(ws + 2949120);
    float*    att   = (float*)(ws + 3538944);
    float*    ybuf  = (float*)(ws + 4718592);
    _Float16* g16   = (_Float16*)(ws + 5898240);
    _Float16* wqh   = (_Float16*)(ws + 8257536);
    _Float16* wkh   = (_Float16*)(ws + 8290304);
    _Float16* wvh   = (_Float16*)(ws + 8323072);
    _Float16* w1h   = (_Float16*)(ws + 8355840);
    _Float16* w2h   = (_Float16*)(ws + 8486912);
    float*    aux   = (float*)(ws + 8617984);
    float*    fused = (float*)(ws + 8691712);
    float*    h1    = (float*)(ws + 9084928);

    const size_t attn_smem = (size_t)16 * SROW * 4 + 128*4 + 256*4 + 16*4 + 16*4 + 512*4;

    k_embed<<<1152, 256, 0, stream>>>(traj_obs, traj_conf, node_pw, node_pb,
                                      time_emb, src_emb, x, xh);

    for (int l = 0; l < 3; l++) {
        k_prep<<<704, 256, 0, stream>>>(Wq_w, Wk_w, Wv_w, ffn1_w, ffn2_w,
                                        traj_obs, traj_conf, edge_w, l,
                                        wqh, wkh, wvh, w1h, w2h, aux);
        k_qkv<<<432, 256, 0, stream>>>(xh, wqh, wkh, wvh,
                                       Wq_b + l * DIMX, Wk_b + l * DIMX, Wv_b + l * DIMX,
                                       Qh, Kh, VT);
        k_attn<<<576, 256, attn_smem, stream>>>(Qh, Kh, VT, aux, edge_w, edge_b, l, att);
        k_addln<<<NN, 128, 0, stream>>>(x, att, ln1_g + l * DIMX, ln1_b + l * DIMX, x, xh);
        k_ffn1<<<576, 256, 0, stream>>>(xh, w1h, ffn1_b + l * FF, g16);
        k_ffn2<<<144, 256, 0, stream>>>(g16, w2h, ffn2_b + l * DIMX, x, ybuf);
        k_addln<<<NN, 128, 0, stream>>>(ybuf, (const float*)nullptr,
                                        ln2_g + l * DIMX, ln2_b + l * DIMX, x, xh);
    }

    k_fuse<<<384, 256, 0, stream>>>(x, fused);
    k_head1<<<384, 256, 0, stream>>>(fused, out1_w, out1_b, h1);
    k_head2<<<9, 256, 0, stream>>>(h1, out2_w, out2_b, (float*)d_out);
}
